// Generator_3831110828035
// MI455X (gfx1250) — matrix operand order repaired and checked by reading, not compiled, not run on hardware
//
#include <hip/hip_runtime.h>
#include <cstdint>
#include <cmath>

// Problem dims (match reference)
#define EMB 256
#define ZDIM 128
#define HDIM 512
#define VOCAB 32000
#define BATCH 64
#define TSTEPS 64
#define BOS 2

#define KX (ZDIM + EMB + HDIM)   // 896  : gate GEMM K  (x = [z | emb | h])
#define NG (4 * HDIM)            // 2048 : gate GEMM N
#define KY (ZDIM + HDIM)         // 640  : logits GEMM K (y = [z | h])

typedef __bf16 bf16_t;
typedef __attribute__((ext_vector_type(16))) __bf16 v16bf;
typedef __attribute__((ext_vector_type(8)))  __bf16 v8bf;
typedef __attribute__((ext_vector_type(8)))  float  v8f;
typedef __attribute__((ext_vector_type(4)))  unsigned int u32x4;
typedef __attribute__((ext_vector_type(8)))  int i32x8;
typedef __attribute__((ext_vector_type(4)))  int i32x4;

static __device__ __forceinline__ bf16_t f2bf(float x) {
  unsigned u = __builtin_bit_cast(unsigned, x);
  unsigned r = u + 0x7FFFu + ((u >> 16) & 1u);       // round-to-nearest-even
  unsigned short s = (unsigned short)(r >> 16);
  return __builtin_bit_cast(bf16_t, s);
}

union FragU { v16bf v; v8bf h[2]; };

// A fragment (16x32 bf16, ISA 7.12.2): lane m=lane&15, K runs {ks..ks+7, ks+16..ks+23}, ks=8*(lane>=16)
template <typename P>
static __device__ __forceinline__ v16bf load_a_frag(P rowbase, int koff) {
  FragU u;
  u.h[0] = *reinterpret_cast<const v8bf*>(&rowbase[koff]);
  u.h[1] = *reinterpret_cast<const v8bf*>(&rowbase[koff + 16]);
  return u.v;
}
// B fragment (32x16 bf16): lane n=lane&15, K runs {ks..ks+7, ks+16..ks+23}, ks=8*(lane>=16)
static __device__ __forceinline__ v16bf load_b_frag(const bf16_t* rowbase, int koff) {
  FragU u;
  u.h[0] = *reinterpret_cast<const v8bf*>(rowbase + koff);
  u.h[1] = *reinterpret_cast<const v8bf*>(rowbase + koff + 16);
  return u.v;
}

// ---------------- setup kernels ----------------

// Wg[n][k] = k<384 ? W_ih[n][k] : W_hh[n][k-384]   (bf16, [2048 x 896])
__global__ __launch_bounds__(256) void k_conv_wg(const float* __restrict__ Wih,
                                                 const float* __restrict__ Whh,
                                                 bf16_t* __restrict__ Wg) {
  int idx = blockIdx.x * 256 + threadIdx.x;
  if (idx >= NG * KX) return;
  int n = idx / KX, k = idx % KX;
  float v = (k < (ZDIM + EMB)) ? Wih[n * (ZDIM + EMB) + k] : Whh[n * HDIM + (k - (ZDIM + EMB))];
  Wg[idx] = f2bf(v);
}

__global__ __launch_bounds__(256) void k_conv_wfc(const float* __restrict__ Wfc,
                                                  bf16_t* __restrict__ Wb) {
  int idx = blockIdx.x * 256 + threadIdx.x;
  if (idx < VOCAB * KY) Wb[idx] = f2bf(Wfc[idx]);
}

// h=c=0 ; x0 = [z | E[BOS] | 0] bf16 ; y z-half = z bf16 (h-half written each step)
__global__ __launch_bounds__(256) void k_init(const float* __restrict__ z,
                                              const float* __restrict__ E,
                                              bf16_t* __restrict__ x, bf16_t* __restrict__ y,
                                              float* __restrict__ h, float* __restrict__ c) {
  int idx = blockIdx.x * 256 + threadIdx.x;
  if (idx < BATCH * KX) {
    int b = idx / KX, k = idx % KX;
    float v;
    if (k < ZDIM)            v = z[b * ZDIM + k];
    else if (k < ZDIM + EMB) v = E[BOS * EMB + (k - ZDIM)];
    else                     v = 0.0f;
    x[idx] = f2bf(v);
  }
  if (idx < BATCH * HDIM) { h[idx] = 0.0f; c[idx] = 0.0f; }
  if (idx < BATCH * ZDIM) {
    int b = idx / ZDIM, k = idx % ZDIM;
    y[b * KY + k] = f2bf(z[idx]);
  }
}

// ---------------- fused gate GEMM + LSTM pointwise ----------------
// 32 blocks (one per 16-wide h-column tile), 4 waves/block: wave q handles gate
// quadrant q (n0 = q*512 + 16*jt), each wave a 64x16 strip over K=896.
// Low occupancy (~1 wave/SIMD) -> fully software-pipelined A/B fragment loads.
__global__ __launch_bounds__(128) void k_gates_lstm(const bf16_t* __restrict__ x,
                                                    const bf16_t* __restrict__ Wg,
                                                    const float* __restrict__ bih,
                                                    const float* __restrict__ bhh,
                                                    float* __restrict__ h, float* __restrict__ c,
                                                    bf16_t* __restrict__ y) {
  __shared__ float sg[4][BATCH][16];     // [gate][batch][col], 16 KB
  const int lane = threadIdx.x & 31;
  const int q    = threadIdx.x >> 5;     // gate quadrant i/f/g/o
  const int jt   = blockIdx.x;           // h-column tile
  const int n0   = q * HDIM + jt * 16;
  const int lr   = lane & 15;
  const int ksa  = (lane >> 4) << 3;
  const int ksb  = (lane >> 4) << 3;
  const bf16_t* brow = Wg + (size_t)(n0 + lr) * KX;
  const bf16_t* a0 = x + (size_t)(0 * 16 + lr) * KX;
  const bf16_t* a1 = x + (size_t)(1 * 16 + lr) * KX;
  const bf16_t* a2 = x + (size_t)(2 * 16 + lr) * KX;
  const bf16_t* a3 = x + (size_t)(3 * 16 + lr) * KX;
  v8f c0 = {}, c1 = {}, c2 = {}, c3 = {};
  // software pipeline: current fragments in regs, next chunk loads issued before the WMMA group
  v16bf bF  = load_b_frag(brow, ksb);
  v16bf aC0 = load_a_frag(a0, ksa);
  v16bf aC1 = load_a_frag(a1, ksa);
  v16bf aC2 = load_a_frag(a2, ksa);
  v16bf aC3 = load_a_frag(a3, ksa);
#pragma unroll
  for (int kc = 0; kc < KX; kc += 32) {
    v16bf bN = bF, aN0 = aC0, aN1 = aC1, aN2 = aC2, aN3 = aC3;
    if (kc + 32 < KX) {
      bN  = load_b_frag(brow, kc + 32 + ksb);
      aN0 = load_a_frag(a0, kc + 32 + ksa);
      aN1 = load_a_frag(a1, kc + 32 + ksa);
      aN2 = load_a_frag(a2, kc + 32 + ksa);
      aN3 = load_a_frag(a3, kc + 32 + ksa);
    }
    __builtin_prefetch(brow + kc + 192, 0, 3);          // near-scope W prefetch
    c0 = __builtin_amdgcn_wmma_f32_16x16x32_bf16(false, aC0, false, bF, (short)0, c0, false, false);
    c1 = __builtin_amdgcn_wmma_f32_16x16x32_bf16(false, aC1, false, bF, (short)0, c1, false, false);
    c2 = __builtin_amdgcn_wmma_f32_16x16x32_bf16(false, aC2, false, bF, (short)0, c2, false, false);
    c3 = __builtin_amdgcn_wmma_f32_16x16x32_bf16(false, aC3, false, bF, (short)0, c3, false, false);
    bF = bN; aC0 = aN0; aC1 = aN1; aC2 = aN2; aC3 = aN3;
  }
  const int mh = (lane >> 4) << 3;
#pragma unroll
  for (int v = 0; v < 8; ++v) {
    sg[q][0 * 16 + v + mh][lr] = c0[v];
    sg[q][1 * 16 + v + mh][lr] = c1[v];
    sg[q][2 * 16 + v + mh][lr] = c2[v];
    sg[q][3 * 16 + v + mh][lr] = c3[v];
  }
  __syncthreads();
  // pointwise LSTM for this block's 64x16 patch (8 elements/thread)
#pragma unroll
  for (int e = threadIdx.x; e < BATCH * 16; e += 128) {
    const int m = e >> 4, col = e & 15;
    const int jg = jt * 16 + col;
    float gi = sg[0][m][col] + bih[jg]             + bhh[jg];
    float gf = sg[1][m][col] + bih[HDIM + jg]      + bhh[HDIM + jg];
    float gg = sg[2][m][col] + bih[2 * HDIM + jg]  + bhh[2 * HDIM + jg];
    float go = sg[3][m][col] + bih[3 * HDIM + jg]  + bhh[3 * HDIM + jg];
    float si = 1.0f / (1.0f + __expf(-gi));
    float sf = 1.0f / (1.0f + __expf(-gf));
    float so = 1.0f / (1.0f + __expf(-go));
    float tg = tanhf(gg);
    const size_t ci = (size_t)m * HDIM + jg;
    float cn = sf * c[ci] + si * tg;
    float hn = so * tanhf(cn);
    c[ci] = cn;
    h[ci] = hn;
    y[(size_t)m * KY + ZDIM + jg] = f2bf(hn);   // z-half of y is static
  }
}

// ---------------- logits GEMM: out[b,t,:] = y[64,640] @ Wfc^T + b_fc ----------------
// 2000 N-tiles, one wave -> 64x16 strip, 8 waves/block, 250 blocks.
// y (80KB) is staged once per block into LDS via the Tensor Data Mover
// (one D#-driven DMA, TENSORcnt-tracked) and shared by all 8 waves.
__global__ __launch_bounds__(256) void k_logits_gemm(const bf16_t* __restrict__ y,
                                                     const bf16_t* __restrict__ Wb,
                                                     const float* __restrict__ bfc,
                                                     float* __restrict__ out, int t) {
  __shared__ __attribute__((aligned(64))) bf16_t sy[BATCH * KY];   // 80 KB activation stage
#if __has_builtin(__builtin_amdgcn_tensor_load_to_lds) && __has_builtin(__builtin_amdgcn_s_wait_tensorcnt)
  if (threadIdx.x < 32) {   // wave 0 issues the TDM descriptor (EXEC ignored by tensor ops)
    const unsigned lds_addr = (unsigned)(size_t)(void*)sy;         // flat addr[31:0] == LDS offset
    const unsigned long long ga = (unsigned long long)(size_t)y;
    // D# group0 (ISA 8.3): count=1 | lds_addr | global_addr[56:0] | type=2
    u32x4 g0 = { 1u, lds_addr,
                 (unsigned)(ga & 0xFFFFFFFFu),
                 (unsigned)((ga >> 32) & 0x01FFFFFFu) | (2u << 30) };
    // D# group1 (ISA 8.4): data_size=2B; tensor 640x64, tile 640x64, dim0_stride=640
    i32x8 g1 = { (int)(1u << 16),                       // data_size code 1 (2 bytes)
                 (int)(((unsigned)KY & 0xFFFFu) << 16), // tensor_dim0[15:0] @ bits63:48
                 (int)(((unsigned)BATCH & 0xFFFFu) << 16), // dim0 hi=0 | tensor_dim1[15:0] @95:80
                 (int)((unsigned)KY << 16),             // dim1 hi=0 | tile_dim0 @127:112
                 (int)((unsigned)BATCH & 0xFFFFu),      // tile_dim1 @143:128, tile_dim2=0
                 (int)KY,                               // tensor_dim0_stride[31:0]
                 0, 0 };                                // stride hi, tensor_dim1_stride (unused, 2D)
    i32x4 zz = { 0, 0, 0, 0 };
#if __has_include(<hip/amd_detail/amd_gfx1250_TDM.h>)
    i32x8 z8 = { 0, 0, 0, 0, 0, 0, 0, 0 };
    __builtin_amdgcn_tensor_load_to_lds(g0, g1, zz, zz, z8, 0);    // 6-arg toolchain
#else
    __builtin_amdgcn_tensor_load_to_lds(g0, g1, zz, zz, 0);        // 5-arg toolchain (ROCm 7.2)
#endif
    __builtin_amdgcn_s_wait_tensorcnt(0);
  }
#else
  {
    const v8bf* src = reinterpret_cast<const v8bf*>(y);
    v8bf* dst = reinterpret_cast<v8bf*>(sy);
#pragma unroll
    for (int i = 0; i < (BATCH * KY / 8) / 256; ++i)
      dst[threadIdx.x + i * 256] = src[threadIdx.x + i * 256];
  }
#endif
  __syncthreads();
  const int lane = threadIdx.x & 31;
  const int wave = threadIdx.x >> 5;
  const int n0   = (blockIdx.x * 8 + wave) * 16;
  const int lr   = lane & 15;
  const int ksa  = (lane >> 4) << 3;
  const int ksb  = (lane >> 4) << 3;
  const bf16_t* brow = Wb + (size_t)(n0 + lr) * KY;
  const bf16_t* a0 = sy + (size_t)(0 * 16 + lr) * KY;
  const bf16_t* a1 = sy + (size_t)(1 * 16 + lr) * KY;
  const bf16_t* a2 = sy + (size_t)(2 * 16 + lr) * KY;
  const bf16_t* a3 = sy + (size_t)(3 * 16 + lr) * KY;
  v8f c0 = {}, c1 = {}, c2 = {}, c3 = {};
  v16bf bF = load_b_frag(brow, ksb);
#pragma unroll
  for (int kc = 0; kc < KY; kc += 32) {
    __builtin_prefetch(brow + kc + 160, 0, 3);
    v16bf bN = bF;
    if (kc + 32 < KY) bN = load_b_frag(brow, kc + 32 + ksb);
    v16bf aF0 = load_a_frag(a0, kc + ksa);
    v16bf aF1 = load_a_frag(a1, kc + ksa);
    v16bf aF2 = load_a_frag(a2, kc + ksa);
    v16bf aF3 = load_a_frag(a3, kc + ksa);
    c0 = __builtin_amdgcn_wmma_f32_16x16x32_bf16(false, aF0, false, bF, (short)0, c0, false, false);
    c1 = __builtin_amdgcn_wmma_f32_16x16x32_bf16(false, aF1, false, bF, (short)0, c1, false, false);
    c2 = __builtin_amdgcn_wmma_f32_16x16x32_bf16(false, aF2, false, bF, (short)0, c2, false, false);
    c3 = __builtin_amdgcn_wmma_f32_16x16x32_bf16(false, aF3, false, bF, (short)0, c3, false, false);
    bF = bN;
  }
  const int mh = (lane >> 4) << 3;
  const float bias = bfc[n0 + lr];
#pragma unroll
  for (int v = 0; v < 8; ++v) {
    out[((size_t)(0 * 16 + v + mh) * TSTEPS + t) * VOCAB + n0 + lr] = c0[v] + bias;
    out[((size_t)(1 * 16 + v + mh) * TSTEPS + t) * VOCAB + n0 + lr] = c1[v] + bias;
    out[((size_t)(2 * 16 + v + mh) * TSTEPS + t) * VOCAB + n0 + lr] = c2[v] + bias;
    out[((size_t)(3 * 16 + v + mh) * TSTEPS + t) * VOCAB + n0 + lr] = c3[v] + bias;
  }
}

// ---------------- argmax (first-max) + build next x ----------------
__global__ __launch_bounds__(256) void k_argmax_buildx(const float* __restrict__ out, int t,
                                                       const float* __restrict__ E,
                                                       const float* __restrict__ h,
                                                       bf16_t* __restrict__ x,
                                                       int* __restrict__ idx_out) {
  __shared__ float sv[256];
  __shared__ int   si[256];
  __shared__ int   stok;
  const int b = blockIdx.x;
  const int tid = threadIdx.x;
  const float* row = out + ((size_t)b * TSTEPS + t) * VOCAB;
  float best = -INFINITY; int bi = VOCAB;
  for (int v = tid; v < VOCAB; v += 256) {
    float xv = row[v];
    if (xv > best || (xv == best && v < bi)) { best = xv; bi = v; }
  }
  sv[tid] = best; si[tid] = bi;
  __syncthreads();
  for (int s = 128; s > 0; s >>= 1) {
    if (tid < s) {
      float ov = sv[tid + s]; int oi = si[tid + s];
      if (ov > sv[tid] || (ov == sv[tid] && oi < si[tid])) { sv[tid] = ov; si[tid] = oi; }
    }
    __syncthreads();
  }
  if (tid == 0) { stok = si[0]; idx_out[b * TSTEPS + t] = si[0]; }
  __syncthreads();
  const int tok = stok;
  // x[b] = [ z (static) | E[tok] | h ]
  for (int k = tid; k < EMB + HDIM; k += 256) {
    if (k < EMB) x[(size_t)b * KX + ZDIM + k] = f2bf(E[(size_t)tok * EMB + k]);
    else         x[(size_t)b * KX + ZDIM + EMB + (k - EMB)] = f2bf(h[(size_t)b * HDIM + (k - EMB)]);
  }
}

// ---------------- launcher ----------------
extern "C" void kernel_launch(void* const* d_in, const int* in_sizes, int n_in,
                              void* d_out, int out_size, void* d_ws, size_t ws_size,
                              hipStream_t stream) {
  const float* z   = (const float*)d_in[0];
  const float* E   = (const float*)d_in[1];
  const float* Wih = (const float*)d_in[2];
  const float* Whh = (const float*)d_in[3];
  const float* bih = (const float*)d_in[4];
  const float* bhh = (const float*)d_in[5];
  const float* Wfc = (const float*)d_in[6];
  const float* bfc = (const float*)d_in[7];
  float* out = (float*)d_out;
  int* idx_out = (int*)(out + (size_t)BATCH * TSTEPS * VOCAB);

  // workspace layout (256B aligned slices)
  char* w = (char*)d_ws;
  size_t off = 0;
  auto take = [&](size_t bytes) { void* p = w + off; off = (off + bytes + 255) & ~(size_t)255; return p; };
  bf16_t* Wg    = (bf16_t*)take((size_t)NG * KX * sizeof(bf16_t));       // 3.67 MB
  bf16_t* Wfcb  = (bf16_t*)take((size_t)VOCAB * KY * sizeof(bf16_t));    // 41 MB (L2-resident)
  bf16_t* x     = (bf16_t*)take((size_t)BATCH * KX * sizeof(bf16_t));
  bf16_t* y     = (bf16_t*)take((size_t)BATCH * KY * sizeof(bf16_t));
  float*  h     = (float*)take((size_t)BATCH * HDIM * sizeof(float));
  float*  c     = (float*)take((size_t)BATCH * HDIM * sizeof(float));
  (void)ws_size; (void)in_sizes; (void)n_in; (void)out_size;

  k_conv_wg <<<(NG * KX + 255) / 256, 256, 0, stream>>>(Wih, Whh, Wg);
  k_conv_wfc<<<((size_t)VOCAB * KY + 255) / 256, 256, 0, stream>>>(Wfc, Wfcb);
  k_init    <<<(BATCH * KX + 255) / 256, 256, 0, stream>>>(z, E, x, y, h, c);

  for (int t = 0; t < TSTEPS; ++t) {
    k_gates_lstm   <<<HDIM / 16, 128, 0, stream>>>(x, Wg, bih, bhh, h, c, y);
    k_logits_gemm  <<<VOCAB / 16 / 8, 256, 0, stream>>>(y, Wfcb, bfc, out, t);
    k_argmax_buildx<<<BATCH, 256, 0, stream>>>(out, t, E, h, x, idx_out);
  }
}
